// Attention_78907139162617
// MI455X (gfx1250) — matrix operand order repaired and checked by reading, not compiled, not run on hardware
//
#include <hip/hip_runtime.h>
#include <hip/hip_bf16.h>

// ---------------------------------------------------------------------------
// Multi-head attention for MI455X (gfx1250), wave32 + WMMA f16 (f32 accum).
// B=4, S=2048, D=1024, H=16, DK=DV=64.
// qkv_proj (x3) -> flash attention -> out_proj, all via v_wmma_f32_16x16x32_f16.
// Attention K tiles staged by the Tensor Data Mover (tensor_load_to_lds,
// TENSORcnt) with hardware LDS padding; out_proj A tiles staged with
// global_load_async_to_lds_b128 (ASYNCcnt); f32->f16 via v_cvt_pk_rtz_f16_f32.
// ---------------------------------------------------------------------------

#define BB   4
#define SS   2048
#define DD   1024
#define HH   16
#define DKV  64

typedef __attribute__((ext_vector_type(16))) _Float16 v16h;
typedef __attribute__((ext_vector_type(8)))  float    v8f;
typedef __attribute__((ext_vector_type(2)))  __fp16   fp16x2;
typedef uint32_t u32x4 __attribute__((ext_vector_type(4)));
typedef uint32_t u32x8 __attribute__((ext_vector_type(8)));

#define WMMA_F16(a, b, c) \
    __builtin_amdgcn_wmma_f32_16x16x32_f16(false, (a), false, (b), (short)0, (c), false, false)

// ---- LDS address helper ---------------------------------------------------
typedef __attribute__((address_space(3))) const void lds_cv;
__device__ __forceinline__ uint32_t lds_off(const void* p) {
    return (uint32_t)(unsigned long long)(lds_cv*)p;
}

// ---- CDNA5 async global->LDS copy (ASYNCcnt) ------------------------------
__device__ __forceinline__ void async_copy_b128(uint32_t lds_addr, const void* gptr) {
    asm volatile("global_load_async_to_lds_b128 %0, %1, off"
                 :: "v"(lds_addr), "v"(gptr) : "memory");
}
__device__ __forceinline__ void wait_async_all() {
    asm volatile("s_wait_asynccnt 0x0" ::: "memory");
}

// ---- CDNA5 Tensor Data Mover: 1D/2D tile -> LDS (TENSORcnt) ---------------
// D# per ISA 8.3/8.4. data_size=2B. Optional LDS padding: pad_amt DWORD-code
// inserted every (1<<pad_int)*2 DWORDs, reproducing padded LDS row strides.
__device__ __forceinline__ void tdm_load_to_lds(uint32_t lds_byte,
                                                const void* gaddr,
                                                uint32_t tile_d0,        // elems
                                                uint32_t tile_d1,        // rows (0 = 1D)
                                                uint32_t tensor_stride0, // elems
                                                uint32_t pad_int,        // code: 0=2,1=4,..DWORDs
                                                uint32_t pad_amt,        // code: n -> n+1 DWORDs
                                                bool     pad_en)
{
    uint64_t ga = (uint64_t)(unsigned long long)gaddr;
    u32x4 g0;
    g0[0] = 1u;                                              // count=1, user mode
    g0[1] = lds_byte;                                        // lds_addr
    g0[2] = (uint32_t)ga;                                    // global_addr[31:0]
    g0[3] = (uint32_t)((ga >> 32) & 0x01FFFFFFu) | (2u << 30); // addr[56:32], type=2
    u32x8 g1;
    g1[0] = (1u << 16)                                       // data_size = 2B
          | ((pad_en ? 1u : 0u) << 20)
          | (pad_int << 22) | (pad_amt << 25);
    const uint32_t td0 = 0x00400000u;                        // generous tensor dims
    const uint32_t td1 = 0x00010000u;                        //  (tiles always in-bounds)
    g1[1] = (td0 & 0xFFFFu) << 16;                           // barrier_addr=0 | td0 lo
    g1[2] = (td0 >> 16) | ((td1 & 0xFFFFu) << 16);           // td0 hi | td1 lo
    g1[3] = (td1 >> 16) | (tile_d0 << 16);                   // td1 hi | tile_dim0
    g1[4] = tile_d1 & 0xFFFFu;                               // tile_dim1 | tile_dim2=0
    g1[5] = tensor_stride0;                                  // dim0 stride lo
    g1[6] = 0u;                                              // stride hi | dim1 stride
    g1[7] = 0u;
    asm volatile("tensor_load_to_lds %0, %1" :: "s"(g0), "s"(g1) : "memory");
}

// ---- packed f32x2 -> f16x2 convert ---------------------------------------
__device__ __forceinline__ uint32_t pack2(float a, float b) {
    union { fp16x2 v; uint32_t u; } u;
    u.v = __builtin_amdgcn_cvt_pkrtz(a, b);
    return u.u;
}
__device__ __forceinline__ uint64_t pack4(float a, float b, float c, float d) {
    return (uint64_t)pack2(a, b) | ((uint64_t)pack2(c, d) << 32);
}

// ---------------------------------------------------------------------------
// Fragment builders per CDNA5 ISA 7.12.2 (wave32).
// A 16x32 f16: lane -> m=L%16, h=L/16; elem e: j=e/2,p=e%2,
//   kBase=(j<4)?2j:16+2(j-4), k=kBase+8h+p
// B 32x16 f16 (from transposed [n][k] buffer): lane -> n=L%16, k=(e<8)?8h+e:16+8h+(e-8)
// C 16x16 f32: VGPR r, lane -> (m=r+8*(L/16), n=L%16)
// ---------------------------------------------------------------------------
__device__ __forceinline__ v16h load_afrag(const _Float16* base, int stride, int lane) {
    int m = lane & 15, h = lane >> 4;
    const _Float16* row = base + m * stride + 8 * h;
    union { v16h v; _Float16 e[16]; } u;
#pragma unroll
    for (int j = 0; j < 8; ++j) {
        int kBase = (j < 4) ? (2 * j) : (16 + 2 * (j - 4));
        u.e[2 * j]     = row[kBase];
        u.e[2 * j + 1] = row[kBase + 1];
    }
    return u.v;
}

__device__ __forceinline__ v16h load_bfrag(const _Float16* baseT, int stride, int lane) {
    int n = lane & 15, h = lane >> 4;
    const _Float16* p = baseT + n * stride + 8 * h;
    union { v16h v; _Float16 e[16]; } u;
#pragma unroll
    for (int e = 0; e < 16; ++e) u.e[e] = p[(e < 8) ? e : (e + 8)];
    return u.v;
}

// ---------------------------------------------------------------------------
// Kernel 1: per-head QKV projection GEMM.  M=8192, N=1024, K=1024.
// 256 threads (8 waves), macro tile 128(M) x 64(N), K-step 32.
// ---------------------------------------------------------------------------
#define AST 40   // LDS stride (halves): 80B rows, 16B-aligned
#define BST 40

__global__ __launch_bounds__(256)
void qkv_proj_kernel(const float* __restrict__ X,     // [B*S, D]
                     const float* __restrict__ W,     // [H, D, 64]
                     const float* __restrict__ bias,  // [H*64]
                     _Float16*    __restrict__ Out)   // [B, H, S, 64]
{
    __shared__ _Float16 As[128 * AST];
    __shared__ _Float16 Bs[64 * BST];

    const int t = threadIdx.x, lane = t & 31, wave = t >> 5;
    const int m0 = blockIdx.x * 128;
    const int n0 = blockIdx.y * 64;
    const int head = blockIdx.y;

    v8f acc[4] = {};

    for (int k0 = 0; k0 < DD; k0 += 32) {
        for (int i = t; i < (128 * 32) / 4; i += 256) {
            int r = i >> 3, c4 = (i & 7) * 4;
            float4 x = *(const float4*)&X[(size_t)(m0 + r) * DD + k0 + c4];
            *(uint64_t*)&As[r * AST + c4] = pack4(x.x, x.y, x.z, x.w);
        }
        for (int i = t; i < (32 * 64) / 4; i += 256) {
            int kk = i >> 4, nn4 = (i & 15) * 4;
            float4 w = *(const float4*)&W[((size_t)head * DD + k0 + kk) * 64 + nn4];
            Bs[(nn4 + 0) * BST + kk] = (_Float16)w.x;
            Bs[(nn4 + 1) * BST + kk] = (_Float16)w.y;
            Bs[(nn4 + 2) * BST + kk] = (_Float16)w.z;
            Bs[(nn4 + 3) * BST + kk] = (_Float16)w.w;
        }
        __syncthreads();

        v16h a = load_afrag(As + (wave * 16) * AST, AST, lane);
#pragma unroll
        for (int j = 0; j < 4; ++j) {
            v16h b = load_bfrag(Bs + (j * 16) * BST, BST, lane);
            acc[j] = WMMA_F16(a, b, acc[j]);
        }
        __syncthreads();
    }

    const int nlo = lane & 15, h = lane >> 4;
#pragma unroll
    for (int j = 0; j < 4; ++j) {
        float bb = bias[n0 + j * 16 + nlo];
#pragma unroll
        for (int r = 0; r < 8; ++r) {
            int m = m0 + wave * 16 + r + 8 * h;
            int b = m >> 11;
            int s = m & (SS - 1);
            size_t addr = (((size_t)b * HH + head) * SS + s) * DKV + (j * 16 + nlo);
            Out[addr] = (_Float16)(acc[j][r] + bb);
        }
    }
}

// ---------------------------------------------------------------------------
// Kernel 2: flash attention, per (b, h, 64 query rows), 128 threads / 4 waves.
// Double-buffered K/V tiles; K staged by the TDM (1D tile + HW padding to
// reproduce KST row stride), V staged with a register transpose.
// Reference applies NO 1/sqrt(dk) scale (faithful to source).
// ---------------------------------------------------------------------------
#define KST 72   // [key 32][dk 64]+pad: 144B rows; 32 DW data + 4 DW HW pad
#define VST 40   // [dv 64][key 32]+pad (transposed V)
#define PST 40   // per-wave [m 16][key 32]+pad

__global__ __launch_bounds__(128)
void attn_kernel(const _Float16* __restrict__ Q,   // [B,H,S,64]
                 const _Float16* __restrict__ K,
                 const _Float16* __restrict__ V,
                 _Float16*       __restrict__ Z)   // [B*S, H*64]
{
    __shared__ _Float16 Ks[2][32 * KST];
    __shared__ _Float16 Vs[2][64 * VST];
    __shared__ _Float16 Ps[4 * 16 * PST];

    const int t = threadIdx.x, lane = t & 31, wave = t >> 5;
    const int q0 = blockIdx.x * 64;
    const int bh = blockIdx.y;
    const _Float16* Qbh = Q + (size_t)bh * SS * DKV;
    const _Float16* Kbh = K + (size_t)bh * SS * DKV;
    const _Float16* Vbh = V + (size_t)bh * SS * DKV;

    // K block (32x64 f16, globally contiguous) via one TDM descriptor:
    // 2048 elems, pad 4 DWORDs (code 3) every 32 DWORDs (code 4) -> KST=72.
    auto stageK = [&](int kb, int buf) {
        if (wave == 0)
            tdm_load_to_lds(lds_off(&Ks[buf][0]), Kbh + (size_t)kb * DKV,
                            /*tile_d0=*/32 * DKV, /*tile_d1=*/0,
                            /*stride0=*/32 * DKV, /*pad_int=*/4, /*pad_amt=*/3, true);
    };
    // V block transposed [dv][key]: b64 global read, 4x b16 LDS scatter
    auto stageV = [&](int kb, int buf) {
        for (int i = t; i < (32 * 64) / 4; i += 128) {
            int key = i >> 4, dv4 = (i & 15) * 4;
            union { uint64_t u; _Float16 h[4]; } vv;
            vv.u = *(const uint64_t*)&Vbh[(size_t)(kb + key) * DKV + dv4];
            Vs[buf][(dv4 + 0) * VST + key] = vv.h[0];
            Vs[buf][(dv4 + 1) * VST + key] = vv.h[1];
            Vs[buf][(dv4 + 2) * VST + key] = vv.h[2];
            Vs[buf][(dv4 + 3) * VST + key] = vv.h[3];
        }
    };

    // Preload Q strip as two A-fragments (dk 0-31, 32-63)
    v16h aQ[2];
    {
        int m = lane & 15, h = lane >> 4;
        const _Float16* qrow = Qbh + (size_t)(q0 + wave * 16 + m) * DKV;
#pragma unroll
        for (int ks = 0; ks < 2; ++ks) {
            union { v16h v; _Float16 e[16]; } u;
#pragma unroll
            for (int j = 0; j < 8; ++j) {
                int kBase = (j < 4) ? (2 * j) : (16 + 2 * (j - 4));
                int k = ks * 32 + kBase + 8 * h;
                u.e[2 * j]     = qrow[k];
                u.e[2 * j + 1] = qrow[k + 1];
            }
            aQ[ks] = u.v;
        }
    }

    float rowm[8], rowl[8];
#pragma unroll
    for (int r = 0; r < 8; ++r) { rowm[r] = -3.0e38f; rowl[r] = 0.0f; }
    v8f o[4] = {};

    // Prologue: stage block 0 into buffer 0
    stageK(0, 0);
    stageV(0, 0);
    if (wave == 0) __builtin_amdgcn_s_wait_tensorcnt(0);
    __syncthreads();

    int cur = 0;
    for (int kb = 0; kb < SS; kb += 32, cur ^= 1) {
        // Stage next block into the other buffer while computing this one
        if (kb + 32 < SS) {
            stageK(kb + 32, cur ^ 1);
            stageV(kb + 32, cur ^ 1);
        }
        const _Float16* Kc = Ks[cur];
        const _Float16* Vc = Vs[cur];

        // Scores: Q(16x64) . K^T(64x32) -> two 16x16 tiles, 2 K-steps
        v8f s0 = {}, s1 = {};
#pragma unroll
        for (int ks = 0; ks < 2; ++ks) {
            v16h b0 = load_bfrag(Kc + 0  * KST + ks * 32, KST, lane);
            v16h b1 = load_bfrag(Kc + 16 * KST + ks * 32, KST, lane);
            s0 = WMMA_F16(aQ[ks], b0, s0);
            s1 = WMMA_F16(aQ[ks], b1, s1);
        }

        // Online softmax; row m=r+8*(L/16) reduces across 16-lane groups
        float scale[8];
#pragma unroll
        for (int r = 0; r < 8; ++r) {
            float mx = fmaxf(s0[r], s1[r]);
#pragma unroll
            for (int off = 1; off < 16; off <<= 1)
                mx = fmaxf(mx, __shfl_xor(mx, off, 32));
            float mnew = fmaxf(rowm[r], mx);
            scale[r] = __expf(rowm[r] - mnew);
            rowm[r] = mnew;
            float p0 = __expf(s0[r] - mnew);
            float p1 = __expf(s1[r] - mnew);
            s0[r] = p0; s1[r] = p1;
            float rs = p0 + p1;
#pragma unroll
            for (int off = 1; off < 16; off <<= 1)
                rs += __shfl_xor(rs, off, 32);
            rowl[r] = rowl[r] * scale[r] + rs;
        }
#pragma unroll
        for (int j = 0; j < 4; ++j)
#pragma unroll
            for (int r = 0; r < 8; ++r) o[j][r] *= scale[r];

        // P (C layout) -> wave-private LDS -> A-fragment layout.
        // Same-wave LDS ordering (s_wait_dscnt) suffices; no barrier needed.
        _Float16* Pw = Ps + wave * 16 * PST;
        {
            int nlo = lane & 15, h = lane >> 4;
#pragma unroll
            for (int r = 0; r < 8; ++r) {
                int m = r + 8 * h;
                Pw[m * PST + nlo]      = (_Float16)s0[r];
                Pw[m * PST + 16 + nlo] = (_Float16)s1[r];
            }
        }
        v16h aP = load_afrag(Pw, PST, lane);
#pragma unroll
        for (int j = 0; j < 4; ++j) {
            v16h bV = load_bfrag(Vc + (j * 16) * VST, VST, lane);
            o[j] = WMMA_F16(aP, bV, o[j]);
        }

        // Next buffer fully staged + this buffer fully consumed
        if (wave == 0) __builtin_amdgcn_s_wait_tensorcnt(0);
        __syncthreads();
    }

    const int nlo = lane & 15, h = lane >> 4;
    const int b = bh >> 4, head = bh & 15;
#pragma unroll
    for (int r = 0; r < 8; ++r) {
        float inv = 1.0f / rowl[r];
        int s = q0 + wave * 16 + r + 8 * h;
#pragma unroll
        for (int j = 0; j < 4; ++j) {
            size_t addr = ((size_t)(b * SS + s)) * (HH * DKV) + head * DKV + j * 16 + nlo;
            Z[addr] = (_Float16)(o[j][r] * inv);
        }
    }
}

// ---------------------------------------------------------------------------
// Kernel 3: output projection. Z(f16) @ Wo(f32->f16) + bo -> f32.
// 256 threads, macro tile 128(M) x 64(N); A staged via async b128 copies.
// ---------------------------------------------------------------------------
__global__ __launch_bounds__(256)
void out_proj_kernel(const _Float16* __restrict__ Zin,  // [B*S, H*64]
                     const float*    __restrict__ Wo,   // [H*64, D]
                     const float*    __restrict__ bo,   // [D]
                     float*          __restrict__ Out)  // [B*S, D]
{
    __shared__ _Float16 As[128 * AST];
    __shared__ _Float16 Bs[64 * BST];

    const int t = threadIdx.x, lane = t & 31, wave = t >> 5;
    const int m0 = blockIdx.x * 128;
    const int n0 = blockIdx.y * 64;

    v8f acc[4] = {};

    for (int k0 = 0; k0 < HH * DKV; k0 += 32) {
        for (int i = t; i < (128 * 32) / 8; i += 256) {
            int r = i >> 2, c8 = (i & 3) * 8;
            async_copy_b128(lds_off(&As[r * AST + c8]),
                            &Zin[(size_t)(m0 + r) * (HH * DKV) + k0 + c8]);
        }
        for (int i = t; i < (32 * 64) / 4; i += 256) {
            int kk = i >> 4, nn4 = (i & 15) * 4;
            float4 w = *(const float4*)&Wo[(size_t)(k0 + kk) * DD + n0 + nn4];
            Bs[(nn4 + 0) * BST + kk] = (_Float16)w.x;
            Bs[(nn4 + 1) * BST + kk] = (_Float16)w.y;
            Bs[(nn4 + 2) * BST + kk] = (_Float16)w.z;
            Bs[(nn4 + 3) * BST + kk] = (_Float16)w.w;
        }
        wait_async_all();
        __syncthreads();

        v16h a = load_afrag(As + (wave * 16) * AST, AST, lane);
#pragma unroll
        for (int j = 0; j < 4; ++j) {
            v16h b = load_bfrag(Bs + (j * 16) * BST, BST, lane);
            acc[j] = WMMA_F16(a, b, acc[j]);
        }
        __syncthreads();
    }

    const int nlo = lane & 15, h = lane >> 4;
#pragma unroll
    for (int j = 0; j < 4; ++j) {
        float bb = bo[n0 + j * 16 + nlo];
#pragma unroll
        for (int r = 0; r < 8; ++r) {
            int m = m0 + wave * 16 + r + 8 * h;
            Out[(size_t)m * DD + n0 + j * 16 + nlo] = acc[j][r] + bb;
        }
    }
}

// ---------------------------------------------------------------------------
// Host launcher
// ---------------------------------------------------------------------------
extern "C" void kernel_launch(void* const* d_in, const int* in_sizes, int n_in,
                              void* d_out, int out_size, void* d_ws, size_t ws_size,
                              hipStream_t stream) {
    (void)in_sizes; (void)n_in; (void)out_size; (void)ws_size;

    const float* Xq = (const float*)d_in[0];
    const float* Xk = (const float*)d_in[1];
    const float* Xv = (const float*)d_in[2];
    const float* Wq = (const float*)d_in[3];
    const float* bq = (const float*)d_in[4];
    const float* Wk = (const float*)d_in[5];
    const float* bk = (const float*)d_in[6];
    const float* Wv = (const float*)d_in[7];
    const float* bv = (const float*)d_in[8];
    const float* Wo = (const float*)d_in[9];
    const float* bo = (const float*)d_in[10];
    float* out = (float*)d_out;

    const size_t per = (size_t)BB * HH * SS * DKV;
    _Float16* qws = (_Float16*)d_ws;
    _Float16* kws = qws + per;
    _Float16* vws = kws + per;
    _Float16* zws = vws + per;

    dim3 gproj((BB * SS) / 128, (HH * DKV) / 64);    // 64 x 16
    qkv_proj_kernel<<<gproj, dim3(256), 0, stream>>>(Xq, Wq, bq, qws);
    qkv_proj_kernel<<<gproj, dim3(256), 0, stream>>>(Xk, Wk, bk, kws);
    qkv_proj_kernel<<<gproj, dim3(256), 0, stream>>>(Xv, Wv, bv, vws);

    dim3 gattn(SS / 64, BB * HH);                    // 32 x 64
    attn_kernel<<<gattn, dim3(128), 0, stream>>>(qws, kws, vws, zws);

    dim3 gout((BB * SS) / 128, DD / 64);             // 64 x 16
    out_proj_kernel<<<gout, dim3(256), 0, stream>>>(zws, Wo, bo, out);
}
